// RWKV_TimeMix_x051a_3178275799150
// MI455X (gfx1250) — matrix operand order repaired and checked by reading, not compiled, not run on hardware
//
#include <hip/hip_runtime.h>
#include <hip/hip_bf16.h>
#include <cstdint>
#include <cstddef>

// ---------------- problem constants ----------------
#define BB 4
#define TT 2048
#define CC 2048
#define HH 32
#define NN 64
#define QQ 256
#define NCHUNK 8
#define EPS_GN 6.4e-4f   // 1e-5 * 64

typedef __attribute__((ext_vector_type(16))) __bf16 v16bf;
typedef __attribute__((ext_vector_type(8)))  float  v8f;

__device__ __forceinline__ v16bf frag_ld(const __bf16* p0, const __bf16* p1) {
  union { v16bf v; uint4 u[2]; } f;
  f.u[0] = *reinterpret_cast<const uint4*>(p0);
  f.u[1] = *reinterpret_cast<const uint4*>(p1);
  return f.v;
}

__device__ __forceinline__ v8f v8f_zero() {
  v8f z;
  #pragma unroll
  for (int i = 0; i < 8; ++i) z[i] = 0.0f;
  return z;
}

// Per-lane async 16-byte copy: global -> LDS (rides ASYNCcnt, no VGPR bounce)
__device__ __forceinline__ void async_copy_b128(const __bf16* gsrc, __bf16* ldst) {
  uint64_t ga = (uint64_t)(uintptr_t)gsrc;
  uint32_t la = (uint32_t)(uintptr_t)ldst;   // low 32 bits of generic addr = LDS byte addr
  asm volatile("global_load_async_to_lds_b128 %0, %1, off"
               :: "v"(la), "v"(ga) : "memory");
}
__device__ __forceinline__ void async_wait0() {
  asm volatile("s_wait_asynccnt 0x0" ::: "memory");
}

// ---------------- fp32 -> bf16 convert ----------------
__global__ __launch_bounds__(256) void cvt_bf16_kernel(const float* __restrict__ in,
                                                       __bf16* __restrict__ out, size_t n) {
  size_t i = (size_t)blockIdx.x * 256 + threadIdx.x;
  if (i < n) out[i] = (__bf16)in[i];
}

// ---------------- time-shift mixing ----------------
__global__ __launch_bounds__(256) void mix_kernel(const float* __restrict__ x,
    const float* __restrict__ mk, const float* __restrict__ mv,
    const float* __restrict__ mr, const float* __restrict__ mg,
    __bf16* __restrict__ xk, __bf16* __restrict__ xv,
    __bf16* __restrict__ xr, __bf16* __restrict__ xg) {
  size_t idx = (size_t)blockIdx.x * 256 + threadIdx.x;
  if (idx >= (size_t)BB * TT * CC) return;
  int c = (int)(idx % CC);
  size_t bt = idx / CC;
  int t = (int)(bt % TT);
  float xc = x[idx];
  float xp = (t > 0) ? x[idx - CC] : 0.0f;
  float xx = xp - xc;
  xk[idx] = (__bf16)(xc + xx * mk[c]);
  xv[idx] = (__bf16)(xc + xx * mv[c]);
  xr[idx] = (__bf16)(xc + xx * mr[c]);
  xg[idx] = (__bf16)(xc + xx * mg[c]);
}

// ---------------- bf16 WMMA GEMM: out = A @ W.T + bias ----------------
// A: [M,K] bf16 row-major, W: [Nn,K] bf16 row-major (B-frag column n == W row n)
// mode 0: bf16 row-major
// mode 1: silu -> f32
// mode 2: f32 (+bias) row-major
// mode 3: bf16 row-major (out) AND bf16 transposed [B,H,N,T] (out2)
// mode 4: bf16 transposed [B,H,N,T] only (out2)
__global__ __launch_bounds__(256) void gemm_bf16_kernel(
    const __bf16* __restrict__ A, const __bf16* __restrict__ W,
    const float* __restrict__ bias, void* __restrict__ out, void* __restrict__ out2,
    int M, int Nn, int K, int mode) {
  const int lane = threadIdx.x & 31;
  const int wave = threadIdx.x >> 5;
  const int wm = wave >> 2, wn = wave & 3;
  const int lm = lane & 15, lh = lane >> 4;
  const int row0 = blockIdx.x * 128 + wm * 64;
  const int col0 = blockIdx.y * 128 + wn * 32;

  v8f acc[4][2];
  #pragma unroll
  for (int i = 0; i < 4; ++i)
    #pragma unroll
    for (int j = 0; j < 2; ++j) acc[i][j] = v8f_zero();

  const __bf16* Arow[4];
  #pragma unroll
  for (int i = 0; i < 4; ++i) Arow[i] = A + (size_t)(row0 + i * 16 + lm) * K;
  const __bf16* Wrow[2];
  #pragma unroll
  for (int j = 0; j < 2; ++j) Wrow[j] = W + (size_t)(col0 + j * 16 + lm) * K;

  for (int kk = 0; kk < K; kk += 32) {
    if (kk + 128 < K) {  // pull next K-panel toward the WGP (global_prefetch_b8)
      __builtin_prefetch(Arow[0] + kk + 128, 0, 1);
      __builtin_prefetch(Wrow[0] + kk + 128, 0, 1);
    }
    v16bf a[4], b[2];
    #pragma unroll
    for (int i = 0; i < 4; ++i) {
      const __bf16* p = Arow[i] + kk + lh * 16;
      a[i] = frag_ld(p, p + 8);
    }
    #pragma unroll
    for (int j = 0; j < 2; ++j) {
      const __bf16* p = Wrow[j] + kk + lh * 16;
      b[j] = frag_ld(p, p + 8);
    }
    #pragma unroll
    for (int i = 0; i < 4; ++i)
      #pragma unroll
      for (int j = 0; j < 2; ++j)
        acc[i][j] = __builtin_amdgcn_wmma_f32_16x16x32_bf16(
            false, a[i], false, b[j], (short)0, acc[i][j], false, false);
  }

  #pragma unroll
  for (int i = 0; i < 4; ++i)
    #pragma unroll
    for (int j = 0; j < 2; ++j) {
      int col = col0 + j * 16 + lm;
      float bia = bias[col];
      #pragma unroll
      for (int e = 0; e < 8; ++e) {
        int row = row0 + i * 16 + lh * 8 + e;
        size_t off = (size_t)row * Nn + col;
        float val = acc[i][j][e] + bia;
        if (mode == 1) {
          ((float*)out)[off] = val / (1.0f + __expf(-val));  // silu
        } else if (mode == 2) {
          ((float*)out)[off] = val;
        } else {
          if (mode != 4) ((__bf16*)out)[off] = (__bf16)val;
          if (mode >= 3) {
            int bidx = row / TT, t = row % TT;
            int hh = col / NN, n = col % NN;
            size_t offT = (((size_t)bidx * HH + hh) * NN + n) * TT + t;
            ((__bf16*)out2)[offT] = (__bf16)val;
          }
        }
      }
    }
}

// ---------------- chunked WKV attention ----------------
// one workgroup (8 waves) per (b,h); serial over 8 chunks of Q=256
// LDS: kkT[64][264] bf16 | vT[64][264] bf16 | attL[64][264] bf16 |
//      stT[64][72] f32   | stB[64][72] bf16
// kkT/vT are filled via global_load_async_to_lds_b128 from the pre-transposed
// [B,H,N,T] copies of k and v (contiguous rows both ends).
#define KPAD 264
#define SPAD 72
#define ATTN_SMEM ((size_t)(3 * 64 * KPAD * 2 + 64 * SPAD * 4 + 64 * SPAD * 2))

__global__ __launch_bounds__(256) void wkv_attn_kernel(
    const __bf16* __restrict__ r, const __bf16* __restrict__ k,
    const __bf16* __restrict__ kTg, const __bf16* __restrict__ vTg,
    const float* __restrict__ time_decay, const float* __restrict__ time_faaaa,
    float* __restrict__ y) {
  extern __shared__ char smem[];
  __bf16* kkT  = (__bf16*)smem;                // [64][KPAD]
  __bf16* vT   = kkT + 64 * KPAD;              // [64][KPAD]
  __bf16* attL = vT + 64 * KPAD;               // [64][KPAD]
  float*  stT  = (float*)(attL + 64 * KPAD);   // [64][SPAD] stateT[nv][nk] f32
  __bf16* stB  = (__bf16*)(stT + 64 * SPAD);   // [64][SPAD] bf16 copy

  const int tid = threadIdx.x;
  const int lane = tid & 31, wave = tid >> 5;
  const int lm = lane & 15, lh = lane >> 4;
  const int wm = wave >> 2, wn = wave & 3;
  const int b = blockIdx.x / HH, h = blockIdx.x % HH;

  const float w = __expf(-__expf(time_decay[h]));
  const float u = time_faaaa[h];
  const float ws = __powf(w, (float)QQ);

  const __bf16* kTbase = kTg + ((size_t)b * HH + h) * NN * TT;
  const __bf16* vTbase = vTg + ((size_t)b * HH + h) * NN * TT;

  for (int i = tid; i < 64 * SPAD; i += 256) { stT[i] = 0.0f; stB[i] = (__bf16)0.0f; }
  __syncthreads();

  for (int chunk = 0; chunk < NCHUNK; ++chunk) {
    const size_t tbase = (size_t)b * TT + (size_t)chunk * QQ;

    // P0: async-stage transposed k,v chunk tiles [n][q] into LDS
    // 64 rows x 512B each = 2048 x 16B lane-chunks, contiguous at both ends.
    for (int idx = tid; idx < 2048; idx += 256) {
      int n = idx >> 5, c = (idx & 31) * 8;              // element offset in row
      const __bf16* gk = kTbase + (size_t)n * TT + chunk * QQ + c;
      const __bf16* gv = vTbase + (size_t)n * TT + chunk * QQ + c;
      async_copy_b128(gk, kkT + n * KPAD + c);
      async_copy_b128(gv, vT + n * KPAD + c);
    }
    async_wait0();
    __syncthreads();

    // P1: per 64-row panel: att = (r@k^T)*Wm -> LDS; y = att@v + (r@state)*wb
    for (int p = 0; p < 4; ++p) {
      // --- att tiles: wave owns mtiles {2wm,2wm+1} x ntiles {4wn..4wn+3}
      {
        v16bf a[2][2], bfr[4][2];
        #pragma unroll
        for (int im = 0; im < 2; ++im) {
          int mt = wm * 2 + im;
          const __bf16* rp = r + (tbase + p * 64 + mt * 16 + lm) * CC + (size_t)h * NN;
          #pragma unroll
          for (int s = 0; s < 2; ++s) {
            const __bf16* pa = rp + s * 32 + lh * 16;
            a[im][s] = frag_ld(pa, pa + 8);
          }
        }
        #pragma unroll
        for (int jn = 0; jn < 4; ++jn) {
          int qj = wn * 64 + jn * 16 + lm;
          const __bf16* kp = k + (tbase + qj) * CC + (size_t)h * NN;
          #pragma unroll
          for (int s = 0; s < 2; ++s) {
            const __bf16* pb = kp + s * 32 + lh * 16;
            bfr[jn][s] = frag_ld(pb, pb + 8);
          }
        }
        v8f acc[2][4];
        #pragma unroll
        for (int im = 0; im < 2; ++im)
          #pragma unroll
          for (int jn = 0; jn < 4; ++jn) acc[im][jn] = v8f_zero();
        #pragma unroll
        for (int s = 0; s < 2; ++s)
          #pragma unroll
          for (int im = 0; im < 2; ++im)
            #pragma unroll
            for (int jn = 0; jn < 4; ++jn)
              acc[im][jn] = __builtin_amdgcn_wmma_f32_16x16x32_bf16(
                  false, a[im][s], false, bfr[jn][s], (short)0, acc[im][jn], false, false);
        // apply Toeplitz decay mask Wm and stash bf16 panel
        #pragma unroll
        for (int im = 0; im < 2; ++im) {
          int mt = wm * 2 + im;
          #pragma unroll
          for (int jn = 0; jn < 4; ++jn) {
            int qj = wn * 64 + jn * 16 + lm;
            #pragma unroll
            for (int e = 0; e < 8; ++e) {
              int rowL = mt * 16 + lh * 8 + e;
              int d = (p * 64 + rowL) - qj;
              float f = (d > 0) ? __powf(w, (float)(d - 1)) : ((d == 0) ? u : 0.0f);
              attL[rowL * KPAD + qj] = (__bf16)(acc[im][jn][e] * f);
            }
          }
        }
      }
      __syncthreads();

      // --- y tiles: wave owns 2 of the 16 (64x64) output tiles
      #pragma unroll
      for (int tt = 0; tt < 2; ++tt) {
        int tidx = wave * 2 + tt;
        int mt = tidx >> 2, nt = tidx & 3;
        // (r @ state) first, then scale by wb
        v8f accy = v8f_zero();
        const __bf16* rp = r + (tbase + p * 64 + mt * 16 + lm) * CC + (size_t)h * NN;
        #pragma unroll
        for (int s = 0; s < 2; ++s) {
          const __bf16* pa = rp + s * 32 + lh * 16;
          v16bf a = frag_ld(pa, pa + 8);
          const __bf16* pb = stB + (nt * 16 + lm) * SPAD + s * 32 + lh * 16;
          v16bf bs = frag_ld(pb, pb + 8);
          accy = __builtin_amdgcn_wmma_f32_16x16x32_bf16(
              false, a, false, bs, (short)0, accy, false, false);
        }
        #pragma unroll
        for (int e = 0; e < 8; ++e) {
          int rowc = p * 64 + mt * 16 + lh * 8 + e;
          accy[e] *= __powf(w, (float)rowc);  // wb
        }
        // += att @ v   (K = 256)
        #pragma unroll
        for (int s = 0; s < 8; ++s) {
          const __bf16* pa = attL + (mt * 16 + lm) * KPAD + s * 32 + lh * 16;
          v16bf a = frag_ld(pa, pa + 8);
          const __bf16* pb = vT + (nt * 16 + lm) * KPAD + s * 32 + lh * 16;
          v16bf bv = frag_ld(pb, pb + 8);
          accy = __builtin_amdgcn_wmma_f32_16x16x32_bf16(
              false, a, false, bv, (short)0, accy, false, false);
        }
        // store y (f32, [B*T, C] layout)
        #pragma unroll
        for (int e = 0; e < 8; ++e) {
          size_t row = tbase + p * 64 + mt * 16 + lh * 8 + e;
          int c = h * NN + nt * 16 + lm;
          y[row * CC + c] = accy[e];
        }
      }
      __syncthreads();
    }

    // P2: fold wkv into v: vT[n][q] *= w^(Q-1-q)
    for (int idx = tid; idx < NN * QQ; idx += 256) {
      int n = idx >> 8, q = idx & 255;
      float val = (float)vT[n * KPAD + q] * __powf(w, (float)(QQ - 1 - q));
      vT[n * KPAD + q] = (__bf16)val;
    }
    __syncthreads();

    // P3: state = ws*state + kk @ (wkv*v)   (output [nk, nv], K = 256)
    #pragma unroll
    for (int tt = 0; tt < 2; ++tt) {
      int tidx = wave * 2 + tt;
      int mt = tidx >> 2, nt = tidx & 3;
      v8f accs = v8f_zero();
      #pragma unroll
      for (int s = 0; s < 8; ++s) {
        const __bf16* pa = kkT + (mt * 16 + lm) * KPAD + s * 32 + lh * 16;
        v16bf a = frag_ld(pa, pa + 8);
        const __bf16* pb = vT + (nt * 16 + lm) * KPAD + s * 32 + lh * 16;
        v16bf bv = frag_ld(pb, pb + 8);
        accs = __builtin_amdgcn_wmma_f32_16x16x32_bf16(
            false, a, false, bv, (short)0, accs, false, false);
      }
      int nv = nt * 16 + lm;
      int nk0 = mt * 16 + lh * 8;
      #pragma unroll
      for (int e = 0; e < 8; ++e) {
        float old = stT[nv * SPAD + nk0 + e];
        stT[nv * SPAD + nk0 + e] = ws * old + accs[e];
      }
    }
    __syncthreads();
    for (int i = tid; i < NN * NN; i += 256) {
      int nv = i >> 6, nk = i & 63;
      stB[nv * SPAD + nk] = (__bf16)stT[nv * SPAD + nk];
    }
    __syncthreads();
  }
}

// ---------------- GroupNorm(H groups) * gate -> bf16 ----------------
__global__ __launch_bounds__(256) void gnorm_gate_kernel(
    const float* __restrict__ y, const float* __restrict__ g,
    const float* __restrict__ ln_g, const float* __restrict__ ln_b,
    __bf16* __restrict__ outb) {
  int wave = threadIdx.x >> 5, lane = threadIdx.x & 31;
  size_t grp = (size_t)blockIdx.x * 8 + wave;        // B*T*H groups
  size_t bt = grp >> 5;
  int h = (int)(grp & 31);
  size_t base = bt * CC + (size_t)h * NN;
  float v0 = y[base + lane], v1 = y[base + 32 + lane];
  float s = v0 + v1, s2 = v0 * v0 + v1 * v1;
  #pragma unroll
  for (int o = 16; o > 0; o >>= 1) {
    s  += __shfl_xor(s, o, 32);
    s2 += __shfl_xor(s2, o, 32);
  }
  float mu = s * (1.0f / 64.0f);
  float var = s2 * (1.0f / 64.0f) - mu * mu;
  float inv = rsqrtf(var + EPS_GN);
  #pragma unroll
  for (int half = 0; half < 2; ++half) {
    int c = h * NN + half * 32 + lane;
    float val = (y[bt * CC + c] - mu) * inv * ln_g[c] + ln_b[c];
    outb[bt * CC + c] = (__bf16)(val * g[bt * CC + c]);
  }
}

// ---------------- host side ----------------
extern "C" void kernel_launch(void* const* d_in, const int* in_sizes, int n_in,
                              void* d_out, int out_size, void* d_ws, size_t ws_size,
                              hipStream_t stream) {
  const float* x   = (const float*)d_in[0];
  const float* Wr  = (const float*)d_in[1];
  const float* br  = (const float*)d_in[2];
  const float* Wk  = (const float*)d_in[3];
  const float* bk  = (const float*)d_in[4];
  const float* Wv  = (const float*)d_in[5];
  const float* bv  = (const float*)d_in[6];
  const float* Wg  = (const float*)d_in[7];
  const float* bg  = (const float*)d_in[8];
  const float* Wo  = (const float*)d_in[9];
  const float* bo  = (const float*)d_in[10];
  const float* lng = (const float*)d_in[11];
  const float* lnb = (const float*)d_in[12];
  const float* mk  = (const float*)d_in[13];
  const float* mv  = (const float*)d_in[14];
  const float* mr  = (const float*)d_in[15];
  const float* mg  = (const float*)d_in[16];
  const float* td  = (const float*)d_in[17];
  const float* tf  = (const float*)d_in[18];

  const size_t MT  = (size_t)BB * TT;          // 8192
  const size_t WBy = (size_t)CC * CC * 2;      // one bf16 weight matrix
  const size_t XB  = MT * CC * 2;              // one bf16 activation matrix
  const size_t FB  = MT * CC * 4;              // one f32 activation matrix

  char* ws = (char*)d_ws;
  size_t off = 0;
  auto take = [&](size_t bytes) { char* p = ws + off; off = (off + bytes + 255) & ~(size_t)255; return p; };
  __bf16* wr_b = (__bf16*)take(WBy);
  __bf16* wk_b = (__bf16*)take(WBy);
  __bf16* wv_b = (__bf16*)take(WBy);
  __bf16* wg_b = (__bf16*)take(WBy);
  __bf16* wo_b = (__bf16*)take(WBy);
  __bf16* xr_b = (__bf16*)take(XB);
  __bf16* xk_b = (__bf16*)take(XB);
  __bf16* xv_b = (__bf16*)take(XB);
  __bf16* xg_b = (__bf16*)take(XB);
  __bf16* r_b  = (__bf16*)take(XB);
  __bf16* k_b  = (__bf16*)take(XB);     // row-major k (att B-frags)
  __bf16* kT_b = (__bf16*)take(XB);     // [B,H,N,T] k (async LDS staging)
  __bf16* vT_b = (__bf16*)take(XB);     // [B,H,N,T] v (async LDS staging)
  float*  g_f  = (float*)take(FB);
  float*  y_f  = (float*)take(FB);
  __bf16* yn_b = (__bf16*)take(XB);
  (void)ws_size; (void)n_in; (void)in_sizes; (void)out_size;

  // 1) weights -> bf16
  {
    size_t n = (size_t)CC * CC;
    int blocks = (int)((n + 255) / 256);
    cvt_bf16_kernel<<<blocks, 256, 0, stream>>>(Wr, wr_b, n);
    cvt_bf16_kernel<<<blocks, 256, 0, stream>>>(Wk, wk_b, n);
    cvt_bf16_kernel<<<blocks, 256, 0, stream>>>(Wv, wv_b, n);
    cvt_bf16_kernel<<<blocks, 256, 0, stream>>>(Wg, wg_b, n);
    cvt_bf16_kernel<<<blocks, 256, 0, stream>>>(Wo, wo_b, n);
  }

  // 2) time-shift mixing
  {
    size_t n = MT * CC;
    mix_kernel<<<(int)((n + 255) / 256), 256, 0, stream>>>(x, mk, mv, mr, mg,
                                                           xk_b, xv_b, xr_b, xg_b);
  }

  // 3) projections (WMMA bf16)
  dim3 ggrid((unsigned)(MT / 128), (unsigned)(CC / 128));
  gemm_bf16_kernel<<<ggrid, 256, 0, stream>>>(xr_b, wr_b, br, (void*)r_b, nullptr,
                                              (int)MT, CC, CC, 0);
  gemm_bf16_kernel<<<ggrid, 256, 0, stream>>>(xk_b, wk_b, bk, (void*)k_b, (void*)kT_b,
                                              (int)MT, CC, CC, 3);
  gemm_bf16_kernel<<<ggrid, 256, 0, stream>>>(xv_b, wv_b, bv, (void*)vT_b, (void*)vT_b,
                                              (int)MT, CC, CC, 4);
  gemm_bf16_kernel<<<ggrid, 256, 0, stream>>>(xg_b, wg_b, bg, (void*)g_f, nullptr,
                                              (int)MT, CC, CC, 1);

  // 4) chunked WKV attention (126 KB dynamic LDS per WG; WGP has 320 KB)
  (void)hipFuncSetAttribute((const void*)wkv_attn_kernel,
                            hipFuncAttributeMaxDynamicSharedMemorySize, (int)ATTN_SMEM);
  wkv_attn_kernel<<<BB * HH, 256, ATTN_SMEM, stream>>>(r_b, k_b, kT_b, vT_b, td, tf, y_f);

  // 5) GroupNorm * gate -> bf16
  {
    size_t groups = MT * HH;
    gnorm_gate_kernel<<<(int)(groups / 8), 256, 0, stream>>>(y_f, g_f, lng, lnb, yn_b);
  }

  // 6) output projection into d_out (f32)
  gemm_bf16_kernel<<<ggrid, 256, 0, stream>>>(yn_b, wo_b, bo, d_out, nullptr,
                                              (int)MT, CC, CC, 2);
}
